// ConvBlock_14242111554126
// MI455X (gfx1250) — matrix operand order repaired and checked by reading, not compiled, not run on hardware
//
#include <hip/hip_runtime.h>
#include <hip/hip_bf16.h>

// ---------------- problem constants (from reference) ----------------
#define N_NODES  50000
#define DIM      512
#define HDIM     1024      // 2*DIM
#define K_HOPS   4
#define B_GRAPHS 64
#define E_HOP    200000
#define BN_EPS   1e-5f

typedef __attribute__((ext_vector_type(16))) __bf16 v16bf;
typedef __attribute__((ext_vector_type(8)))  float  v8f;

// LDS tile strides (bf16 elements). 40*2 = 80 bytes keeps 16-byte
// alignment for b128 access and spreads the 16 lanes of a fragment
// read over 16 distinct 4-bank groups (20*col mod 64) -> conflict-free.
#define A_STRIDE 40
#define B_STRIDE 40

// ====================================================================
// 1) h0 = X[0] + vn[batch];  result = (1+eps[0]) * h0
// ====================================================================
__global__ __launch_bounds__(256)
void init_h0_result(const float* __restrict__ X0, const float* __restrict__ vn,
                    const int* __restrict__ batch, const float* __restrict__ eps0,
                    float* __restrict__ h0, float* __restrict__ result) {
    int idx = blockIdx.x * 256 + threadIdx.x;   // over N*DIM (exactly divisible)
    int r = idx >> 9;
    int c = idx & (DIM - 1);
    float x = X0[idx] + vn[batch[r] * DIM + c];
    h0[idx] = x;
    result[idx] = (1.0f + eps0[c]) * x;
}

// ====================================================================
// 2) Per-hop edge kernel: result[d] += (1+eps)*relu(x[s] (+ ea))
//    One wave (32 lanes) per edge; each lane covers 16 features.
// ====================================================================
__global__ __launch_bounds__(256)
void edge_scatter(const float* __restrict__ Xsrc,
                  const int* __restrict__ s, const int* __restrict__ d,
                  const int* __restrict__ eattr, const float* __restrict__ ee_tab,
                  const float* __restrict__ eps_row,
                  float* __restrict__ result, int useEa) {
    int wave = threadIdx.x >> 5;
    int lane = threadIdx.x & 31;
    int e = blockIdx.x * 8 + wave;
    if (e >= E_HOP) return;
    int sn = s[e];
    int dn = d[e];
    const float* xj = Xsrc + (size_t)sn * DIM;
    float* out = result + (size_t)dn * DIM;
    const float* ea = useEa ? (ee_tab + (size_t)eattr[e] * DIM) : nullptr;
    int f0 = lane * 16;
#pragma unroll
    for (int j = 0; j < 16; j += 4) {
        float4 x = *(const float4*)(xj + f0 + j);
        if (useEa) {
            float4 a = *(const float4*)(ea + f0 + j);
            x.x += a.x; x.y += a.y; x.z += a.z; x.w += a.w;
        }
        float4 ep = *(const float4*)(eps_row + f0 + j);
        atomicAdd(out + f0 + j + 0, fmaxf(x.x, 0.f) * (1.f + ep.x));
        atomicAdd(out + f0 + j + 1, fmaxf(x.y, 0.f) * (1.f + ep.y));
        atomicAdd(out + f0 + j + 2, fmaxf(x.z, 0.f) * (1.f + ep.z));
        atomicAdd(out + f0 + j + 3, fmaxf(x.w, 0.f) * (1.f + ep.w));
    }
}

// ====================================================================
// 3) bf16 WMMA GEMM:  Z[M x NC] = A[M x KD] * W[KD x NC] + bias
//    Block: 256 threads = 8 waves; tile BM=64, BN=128, BK=32.
//    A tile:  row-major (64 x 32, padded stride 40)  -> frag = 2x ds_load_b128
//    B tile:  TRANSPOSED (128 cols x 32 k, stride 40) -> frag = 2x ds_load_b128
// ====================================================================
__global__ __launch_bounds__(256)
void gemm_bf16(const float* __restrict__ A, const float* __restrict__ W,
               const float* __restrict__ bias, float* __restrict__ Z,
               int M, int KD, int NC) {
    __shared__ __bf16 As[64 * A_STRIDE];    // 5 KB
    __shared__ __bf16 Bt[128 * B_STRIDE];   // 10 KB (col-major / fragment-ready)
    int tid   = threadIdx.x;
    int wave  = tid >> 5;
    int lane  = tid & 31;
    int waveM = wave >> 1;            // 0..3 -> 16-row subtile
    int waveN = wave & 1;             // 0..1 -> 64-col strip
    int half  = lane >> 4;
    int l16   = lane & 15;
    int bm = blockIdx.y * 64;
    int bn = blockIdx.x * 128;

    v8f acc[4] = {};

    for (int k0 = 0; k0 < KD; k0 += 32) {
        // ---- A tile: 64x32 f32 -> bf16, one b128 LDS store per thread
        {
            int row = tid >> 2;               // 0..63
            int cb  = (tid & 3) * 8;          // 0,8,16,24
            int gr  = bm + row; if (gr >= M) gr = M - 1;
            const float* ap = A + (size_t)gr * KD + k0 + cb;
            if (k0 + 32 < KD) __builtin_prefetch(ap + 32, 0, 0);   // global_prefetch_b8
            float4 x0 = *(const float4*)ap;
            float4 x1 = *(const float4*)(ap + 4);
            union { __bf16 h[8]; uint4 u; } pk;
            pk.h[0] = (__bf16)x0.x; pk.h[1] = (__bf16)x0.y;
            pk.h[2] = (__bf16)x0.z; pk.h[3] = (__bf16)x0.w;
            pk.h[4] = (__bf16)x1.x; pk.h[5] = (__bf16)x1.y;
            pk.h[6] = (__bf16)x1.z; pk.h[7] = (__bf16)x1.w;
            *(uint4*)(As + row * A_STRIDE + cb) = pk.u;   // 16B aligned
        }
        // ---- B tile: 32x128 f32, transposed into Bt[col][k].
        // 4x4 micro-blocks: coalesced float4 global loads, b64 LDS stores.
        {
            int kg = tid >> 5;                // 0..7  -> k rows kg*4..+3
            int cg = tid & 31;                // 0..31 -> cols cg*4..+3
            const float* wp = W + (size_t)(k0 + kg * 4) * NC + bn + cg * 4;
            if (k0 + 32 < KD) __builtin_prefetch(wp + (size_t)32 * NC, 0, 0);
            float4 r0 = *(const float4*)(wp);
            float4 r1 = *(const float4*)(wp + NC);
            float4 r2 = *(const float4*)(wp + 2 * (size_t)NC);
            float4 r3 = *(const float4*)(wp + 3 * (size_t)NC);
            union { __bf16 h[4]; uint2 u; } c0, c1, c2, c3;
            c0.h[0]=(__bf16)r0.x; c0.h[1]=(__bf16)r1.x; c0.h[2]=(__bf16)r2.x; c0.h[3]=(__bf16)r3.x;
            c1.h[0]=(__bf16)r0.y; c1.h[1]=(__bf16)r1.y; c1.h[2]=(__bf16)r2.y; c1.h[3]=(__bf16)r3.y;
            c2.h[0]=(__bf16)r0.z; c2.h[1]=(__bf16)r1.z; c2.h[2]=(__bf16)r2.z; c2.h[3]=(__bf16)r3.z;
            c3.h[0]=(__bf16)r0.w; c3.h[1]=(__bf16)r1.w; c3.h[2]=(__bf16)r2.w; c3.h[3]=(__bf16)r3.w;
            __bf16* bp = Bt + (cg * 4) * B_STRIDE + kg * 4;       // 8B aligned
            *(uint2*)(bp)                = c0.u;
            *(uint2*)(bp + B_STRIDE)     = c1.u;
            *(uint2*)(bp + 2 * B_STRIDE) = c2.u;
            *(uint2*)(bp + 3 * B_STRIDE) = c3.u;
        }
        __syncthreads();

        // ---- A fragment (ISA 7.12.2 16-bit A layout):
        // lanes 0-15:  row M=l16, K = 0..7 | 16..23
        // lanes 16-31: row M=l16, K = 8..15 | 24..31
        union { v16bf v; uint4 q[2]; } af;
        {
            const __bf16* ar = As + (waveM * 16 + l16) * A_STRIDE;
            int kb0 = half ? 8 : 0;
            af.q[0] = *(const uint4*)(ar + kb0);        // K kb0..kb0+7
            af.q[1] = *(const uint4*)(ar + kb0 + 16);   // K kb0+16..kb0+23
        }
        // ---- 4 WMMAs; B fragment = same K order as A at this lane's column
        // (lanes 0-15: K 0..7 | 16..23, lanes 16-31: K 8..15 | 24..31, 2 K per VGPR)
        int kb = half ? 8 : 0;
#pragma unroll
        for (int nn = 0; nn < 4; nn++) {
            union { v16bf v; uint4 q[2]; } bfg;
            const __bf16* bp = Bt + (waveN * 64 + nn * 16 + l16) * B_STRIDE + kb;
            bfg.q[0] = *(const uint4*)(bp);
            bfg.q[1] = *(const uint4*)(bp + 16);
            acc[nn] = __builtin_amdgcn_wmma_f32_16x16x32_bf16(
                false, af.v, false, bfg.v, (short)0, acc[nn], false, false);
        }
        __syncthreads();
    }

    // ---- store D (C/D layout: VGPR r -> M = waveM*16 + r + 8*half, N = l16)
#pragma unroll
    for (int nn = 0; nn < 4; nn++) {
        int col = bn + waveN * 64 + nn * 16 + l16;
        float bv = bias[col];
#pragma unroll
        for (int r = 0; r < 8; r++) {
            int grow = bm + waveM * 16 + r + (half ? 8 : 0);
            if (grow < M) Z[(size_t)grow * NC + col] = acc[nn][r] + bv;
        }
    }
}

// ====================================================================
// 4) BatchNorm over M rows: two-pass (partial sums via atomics)
// ====================================================================
__global__ __launch_bounds__(256)
void bn_stats(const float* __restrict__ Z, int M, int C, int rowsPerBlock,
              float* __restrict__ sums) {   // sums[0..C)=sum, [C..2C)=sumsq
    int c = blockIdx.x * 256 + threadIdx.x;
    int r0 = blockIdx.y * rowsPerBlock;
    int r1 = r0 + rowsPerBlock; if (r1 > M) r1 = M;
    float s = 0.f, s2 = 0.f;
    for (int r = r0; r < r1; r++) {
        float x = Z[(size_t)r * C + c];
        s += x; s2 += x * x;
    }
    atomicAdd(&sums[c], s);
    atomicAdd(&sums[C + c], s2);
}

__global__ __launch_bounds__(256)
void bn_finalize(const float* __restrict__ sums, const float* __restrict__ gamma,
                 const float* __restrict__ beta, int C, int M,
                 float* __restrict__ scsh) { // [0..C)=scale, [C..2C)=shift
    int c = blockIdx.x * 256 + threadIdx.x;
    if (c >= C) return;
    float inv = 1.0f / (float)M;
    float mean = sums[c] * inv;
    float var  = sums[C + c] * inv - mean * mean;
    float sc = gamma[c] * rsqrtf(var + BN_EPS);
    scsh[c] = sc;
    scsh[C + c] = beta[c] - mean * sc;
}

// relu(bn(Z)) in place
__global__ __launch_bounds__(256)
void bn_apply_relu(float* __restrict__ Z, const float* __restrict__ scsh,
                   int M, int C) {
    int idx = blockIdx.x * 256 + threadIdx.x;
    int c = idx & (C - 1);
    Z[idx] = fmaxf(Z[idx] * scsh[c] + scsh[C + c], 0.f);
}

// final node output: h -> d_out, and fused batch pooling G += h
__global__ __launch_bounds__(256)
void bn_apply_relu_out(const float* __restrict__ Z, const float* __restrict__ scsh,
                       const int* __restrict__ batch, float* __restrict__ out,
                       float* __restrict__ G, int M, int C) {
    int idx = blockIdx.x * 256 + threadIdx.x;
    int r = idx >> 9;            // C == 512 here
    int c = idx & (C - 1);
    float h = fmaxf(Z[idx] * scsh[c] + scsh[C + c], 0.f);
    out[idx] = h;
    atomicAdd(&G[batch[r] * C + c], h);
}

// ====================================================================
// 5) virtual-node branch (64 rows -- negligible cost, plain VALU)
// ====================================================================
__global__ __launch_bounds__(256)
void vnode_prep(const float* __restrict__ vn, const float* __restrict__ G,
                float* __restrict__ v0) {
    int idx = blockIdx.x * 256 + threadIdx.x;   // B*DIM
    v0[idx] = vn[idx] + G[idx];
}

__global__ __launch_bounds__(256)
void small_gemm(const float* __restrict__ A, const float* __restrict__ W,
                const float* __restrict__ bias, float* __restrict__ Z,
                int M, int KD, int NC) {
    int idx = blockIdx.x * 256 + threadIdx.x;
    if (idx >= M * NC) return;
    int r = idx / NC, c = idx % NC;
    float s = bias[c];
    for (int k = 0; k < KD; k++) s += A[r * KD + k] * W[k * NC + c];
    Z[idx] = s;
}

__global__ __launch_bounds__(256)
void colwise_bn_relu(const float* __restrict__ Z, const float* __restrict__ gamma,
                     const float* __restrict__ beta, float* __restrict__ out,
                     int M, int C) {
    int c = blockIdx.x * 256 + threadIdx.x;
    if (c >= C) return;
    float s = 0.f, s2 = 0.f;
    for (int r = 0; r < M; r++) { float x = Z[r * C + c]; s += x; s2 += x * x; }
    float inv = 1.0f / (float)M;
    float mean = s * inv;
    float var  = s2 * inv - mean * mean;
    float sc = gamma[c] * rsqrtf(var + BN_EPS);
    float sh = beta[c] - mean * sc;
    for (int r = 0; r < M; r++)
        out[r * C + c] = fmaxf(Z[r * C + c] * sc + sh, 0.f);
}

// ====================================================================
// launcher
// ====================================================================
extern "C" void kernel_launch(void* const* d_in, const int* in_sizes, int n_in,
                              void* d_out, int out_size, void* d_ws, size_t ws_size,
                              hipStream_t stream) {
    const float* X    = (const float*)d_in[0];
    const float* vn   = (const float*)d_in[1];
    const float* eps  = (const float*)d_in[2];
    const float* ee   = (const float*)d_in[3];
    const float* w1   = (const float*)d_in[4];
    const float* b1   = (const float*)d_in[5];
    const float* g1   = (const float*)d_in[6];
    const float* be1  = (const float*)d_in[7];
    const float* w2   = (const float*)d_in[8];
    const float* b2   = (const float*)d_in[9];
    const float* gn   = (const float*)d_in[10];
    const float* bnb  = (const float*)d_in[11];
    const float* vw1  = (const float*)d_in[12];
    const float* vb1  = (const float*)d_in[13];
    const float* vg1  = (const float*)d_in[14];
    const float* vbe1 = (const float*)d_in[15];
    const float* vw2  = (const float*)d_in[16];
    const float* vb2  = (const float*)d_in[17];
    const float* vgo  = (const float*)d_in[18];
    const float* vbo  = (const float*)d_in[19];
    const int*   src  = (const int*)d_in[20];
    const int*   dst  = (const int*)d_in[21];
    const int*   batch= (const int*)d_in[22];
    const int*   eatt = (const int*)d_in[23];
    float* out = (float*)d_out;

    // workspace layout (result + h0/Z2 + Z1 dominate: ~410 MB)
    float* resultBuf = (float*)d_ws;                              // N*DIM
    float* h0Buf = resultBuf + (size_t)N_NODES * DIM;             // N*DIM (reused as Z2)
    float* Z1    = h0Buf + (size_t)N_NODES * DIM;                 // N*HDIM
    float* p     = Z1 + (size_t)N_NODES * HDIM;
    float* sums1 = p; p += 2 * HDIM;
    float* scsh1 = p; p += 2 * HDIM;
    float* sums2 = p; p += 2 * DIM;
    float* scsh2 = p; p += 2 * DIM;
    float* G     = p; p += B_GRAPHS * DIM;
    float* v0    = p; p += B_GRAPHS * DIM;
    float* Zv1   = p; p += B_GRAPHS * HDIM;
    float* Hv    = p; p += B_GRAPHS * HDIM;
    float* Zv2   = p; p += B_GRAPHS * DIM;

    hipMemsetAsync(sums1, 0, 2 * HDIM * sizeof(float), stream);
    hipMemsetAsync(sums2, 0, 2 * DIM * sizeof(float), stream);
    hipMemsetAsync(G, 0, B_GRAPHS * DIM * sizeof(float), stream);

    // (1) init
    init_h0_result<<<(N_NODES * DIM) / 256, 256, 0, stream>>>(X, vn, batch, eps,
                                                              h0Buf, resultBuf);
    // (2) 4 hops of gather/relu/scale/scatter-add
    for (int i = 0; i < K_HOPS; i++) {
        const float* Xsrc = (i == 0) ? h0Buf : (X + (size_t)i * N_NODES * DIM);
        edge_scatter<<<E_HOP / 8, 256, 0, stream>>>(
            Xsrc, src + (size_t)i * E_HOP, dst + (size_t)i * E_HOP,
            eatt, ee, eps + (size_t)(i + 1) * DIM, resultBuf, (i == 0) ? 1 : 0);
    }
    // (3) MLP layer 1: Z1 = result @ w1 + b1  (bf16 WMMA)
    gemm_bf16<<<dim3(HDIM / 128, (N_NODES + 63) / 64), 256, 0, stream>>>(
        resultBuf, w1, b1, Z1, N_NODES, DIM, HDIM);
    int rpb = (N_NODES + 127) / 128;
    bn_stats<<<dim3(HDIM / 256, 128), 256, 0, stream>>>(Z1, N_NODES, HDIM, rpb, sums1);
    bn_finalize<<<HDIM / 256, 256, 0, stream>>>(sums1, g1, be1, HDIM, N_NODES, scsh1);
    bn_apply_relu<<<(N_NODES * HDIM) / 256, 256, 0, stream>>>(Z1, scsh1, N_NODES, HDIM);
    // MLP layer 2: Z2 = H1 @ w2 + b2  (bf16 WMMA), Z2 reuses h0 buffer
    gemm_bf16<<<dim3(DIM / 128, (N_NODES + 63) / 64), 256, 0, stream>>>(
        Z1, w2, b2, h0Buf, N_NODES, HDIM, DIM);
    bn_stats<<<dim3(DIM / 256, 128), 256, 0, stream>>>(h0Buf, N_NODES, DIM, rpb, sums2);
    bn_finalize<<<DIM / 256, 256, 0, stream>>>(sums2, gn, bnb, DIM, N_NODES, scsh2);
    // final node output + fused batch pooling into G
    bn_apply_relu_out<<<(N_NODES * DIM) / 256, 256, 0, stream>>>(
        h0Buf, scsh2, batch, out, G, N_NODES, DIM);
    // (4) virtual node branch
    vnode_prep<<<(B_GRAPHS * DIM) / 256, 256, 0, stream>>>(vn, G, v0);
    small_gemm<<<(B_GRAPHS * HDIM) / 256, 256, 0, stream>>>(v0, vw1, vb1, Zv1,
                                                            B_GRAPHS, DIM, HDIM);
    colwise_bn_relu<<<HDIM / 256, 256, 0, stream>>>(Zv1, vg1, vbe1, Hv, B_GRAPHS, HDIM);
    small_gemm<<<(B_GRAPHS * DIM) / 256, 256, 0, stream>>>(Hv, vw2, vb2, Zv2,
                                                           B_GRAPHS, HDIM, DIM);
    colwise_bn_relu<<<DIM / 256, 256, 0, stream>>>(Zv2, vgo, vbo,
                                                   out + (size_t)N_NODES * DIM,
                                                   B_GRAPHS, DIM);
}
